// HicLSTM3D_rep_85615878078996
// MI455X (gfx1250) — compile-verified, matrix operand order checked by reading, not run on hardware
//
#include <hip/hip_runtime.h>

#define B_SZ   2048
#define L_SZ   150
#define E_SZ   16
#define H_SZ   64
#define G_SZ   256      // 4*H
#define LH     9600     // L*H
#define NOUT   150
#define LNEPS  1e-5f

typedef unsigned short u16;
typedef __bf16  bf16_t;
typedef bf16_t  v16bf __attribute__((ext_vector_type(16)));
typedef u16     v16u  __attribute__((ext_vector_type(16)));
typedef float   v8f   __attribute__((ext_vector_type(8)));

static __device__ __forceinline__ v8f wmma_bf16(v16u a, v16u b, v8f c) {
  return __builtin_amdgcn_wmma_f32_16x16x32_bf16(
      false, __builtin_bit_cast(v16bf, a),
      false, __builtin_bit_cast(v16bf, b),
      (short)0, c, false, false);
}

static __device__ __forceinline__ u16 to_bf16(float f) {
  unsigned u = __float_as_uint(f);
  u += 0x7fffu + ((u >> 16) & 1u);   // round-to-nearest-even
  return (u16)(u >> 16);
}

static __device__ __forceinline__ float wsum(float x) {
#pragma unroll
  for (int m = 16; m >= 1; m >>= 1) x += __shfl_xor(x, m, 32);
  return x;
}

// hardware-transcendental-only activations: v_exp_f32 + v_rcp_f32, no IEEE
// divide expansion (v_div_fmas/v_div_fixup) on the serial LSTM chain
static __device__ __forceinline__ float sigmoidf(float x) {
  return __builtin_amdgcn_rcpf(1.0f + __expf(-x));
}
static __device__ __forceinline__ float tanh_fast(float x) {
  // tanh(x) = 2/(1+exp(-2x)) - 1
  return 2.0f * __builtin_amdgcn_rcpf(1.0f + __expf(-2.0f * x)) - 1.0f;
}

// ---------------- f32 -> bf16 conversion ----------------
__global__ void k_cvt_bf16(const float* __restrict__ src, u16* __restrict__ dst, int n) {
  int i = blockIdx.x * blockDim.x + threadIdx.x;
  if (i < n) dst[i] = to_bf16(src[i]);
}

// ---------------- embedding gather + reparameterization ----------------
__global__ void k_reparam(const int* __restrict__ idx, const float* __restrict__ eps,
                          const float* __restrict__ mt, const float* __restrict__ vt,
                          u16* __restrict__ rep) {
  int i = blockIdx.x * blockDim.x + threadIdx.x;    // over B*L
  if (i >= B_SZ * L_SZ) return;
  long r = (long)idx[i] * E_SZ;
  const float* ep = eps + (long)i * E_SZ;
  u16* out = rep + (long)i * E_SZ;
#pragma unroll
  for (int k = 0; k < E_SZ; ++k)
    out[k] = to_bf16(ep[k] * __expf(0.5f * vt[r + k]) + mt[r + k]);
}

// ---------------- layer-normed LSTM over 150 steps ----------------
// 1 block = 16 batch rows, 8 waves; wave w owns gate N-tiles 2w, 2w+1.
__global__ __launch_bounds__(256) void k_lstm(
    const u16* __restrict__ rep, const u16* __restrict__ wih, const u16* __restrict__ whh,
    const float* __restrict__ h0, const float* __restrict__ c0,
    const float* __restrict__ gih_g, const float* __restrict__ gih_b,
    const float* __restrict__ ghh_g, const float* __restrict__ ghh_b,
    const float* __restrict__ gc_g,  const float* __restrict__ gc_b,
    u16* __restrict__ hs) {
  __shared__ float sGI[16][G_SZ];
  __shared__ float sGH[16][G_SZ];
  __shared__ float sC[16][H_SZ];
  __shared__ __attribute__((aligned(32))) u16 sH[16][H_SZ];

  const int tid  = threadIdx.x;
  const int wave = tid >> 5;
  const int lane = tid & 31;
  const int rowb = blockIdx.x * 16;

  // init state
  for (int i = tid; i < 16 * H_SZ; i += 256) {
    int r = i >> 6, j = i & 63;
    sC[r][j] = c0[(long)(rowb + r) * H_SZ + j];
    sH[r][j] = to_bf16(h0[(long)(rowb + r) * H_SZ + j]);
  }

  const int n0    = wave * 2;
  const int halfk = lane >> 4;     // 0: K 0..15, 1: K 16..31 of a k-tile
  const int nlane = lane & 15;

  v16u zero16;
#pragma unroll
  for (int q = 0; q < 16; ++q) zero16[q] = (u16)0;
  v8f zero8 = {0.f, 0.f, 0.f, 0.f, 0.f, 0.f, 0.f, 0.f};

  // resident weight B-fragments (column-per-lane, 16 contiguous K)
  v16u fIH[2], fHH[2][2];
#pragma unroll
  for (int j = 0; j < 2; ++j) {
    int col = (n0 + j) * 16 + nlane;            // gate column 0..255
    fIH[j] = zero16;
    if (halfk == 0)                              // E=16: K 16..31 is zero pad
      fIH[j] = *(const v16u*)(wih + col * E_SZ);
#pragma unroll
    for (int kt = 0; kt < 2; ++kt) {
      int kb = kt * 32 + halfk * 16;
      fHH[j][kt] = *(const v16u*)(whh + col * H_SZ + kb);
    }
  }

  // software-pipelined x fragment: t = 0
  const u16* xbase = rep + (long)(rowb + nlane) * L_SZ * E_SZ;
  v16u ax = zero16;
  if (halfk == 0) ax = *(const v16u*)(xbase);

  __syncthreads();

  for (int t = 0; t < L_SZ; ++t) {
    // ---- h A-fragments from previous step ----
    v16u ah0 = *(const v16u*)(&sH[nlane][halfk * 16]);
    v16u ah1 = *(const v16u*)(&sH[nlane][32 + halfk * 16]);

    // ---- gate matmuls (bf16 WMMA, f32 accum) ----
#pragma unroll
    for (int j = 0; j < 2; ++j) {
      v8f aI = wmma_bf16(ax, fIH[j], zero8);
      v8f aH = wmma_bf16(ah1, fHH[j][1], wmma_bf16(ah0, fHH[j][0], zero8));
      int col = (n0 + j) * 16 + nlane;
      int r0  = halfk * 8;
#pragma unroll
      for (int r = 0; r < 8; ++r) {
        sGI[r0 + r][col] = aI[r];
        sGH[r0 + r][col] = aH[r];
      }
    }

    // prefetch next step's x fragment; its latency overlaps the whole
    // layernorm/elementwise phase instead of sitting on the serial chain
    v16u axn = zero16;
    if (halfk == 0 && (t + 1) < L_SZ)
      axn = *(const v16u*)(xbase + (t + 1) * E_SZ);

    __syncthreads();

    // ---- dual layernorm + gate combine: wave handles rows 2w, 2w+1 ----
#pragma unroll
    for (int rr = 0; rr < 2; ++rr) {
      int r = wave * 2 + rr;
      float vI[8], vH[8];
      float sa = 0.f, qa = 0.f, sb = 0.f, qb = 0.f;
#pragma unroll
      for (int i = 0; i < 8; ++i) {
        float a = sGI[r][lane + 32 * i];
        float b = sGH[r][lane + 32 * i];
        vI[i] = a; vH[i] = b;
        sa += a; qa += a * a; sb += b; qb += b * b;
      }
      sa = wsum(sa); qa = wsum(qa); sb = wsum(sb); qb = wsum(qb);
      float mI = sa * (1.0f / 256.0f);
      float rI = rsqrtf(qa * (1.0f / 256.0f) - mI * mI + LNEPS);
      float mH = sb * (1.0f / 256.0f);
      float rH = rsqrtf(qb * (1.0f / 256.0f) - mH * mH + LNEPS);
#pragma unroll
      for (int i = 0; i < 8; ++i) {
        int col = lane + 32 * i;
        float g = (vI[i] - mI) * rI * gih_g[col] + gih_b[col]
                + (vH[i] - mH) * rH * ghh_g[col] + ghh_b[col];
        sGI[r][col] = g;                 // reuse buffer for combined gates
      }
    }

    // Rows 2w,2w+1 are written and re-read by THIS wave only; per-wave LDS
    // ops complete in order, so only a compiler scheduling fence is needed
    // (saves a full workgroup barrier on the 150-step serial chain).
    __builtin_amdgcn_wave_barrier();

    // ---- cell/hidden update + cell layernorm ----
#pragma unroll
    for (int rr = 0; rr < 2; ++rr) {
      int r = wave * 2 + rr;
      float cv[2], ov[2];
      float s = 0.f, q = 0.f;
#pragma unroll
      for (int k = 0; k < 2; ++k) {
        int j = lane * 2 + k;
        float ig = sGI[r][j];
        float fg = sGI[r][64 + j];
        float gg = sGI[r][128 + j];
        float og = sGI[r][192 + j];
        float c  = sigmoidf(fg) * sC[r][j] + sigmoidf(ig) * tanh_fast(gg);
        sC[r][j] = c;
        cv[k] = c; ov[k] = og;
        s += c; q += c * c;
      }
      s = wsum(s); q = wsum(q);
      float mc = s * (1.0f / 64.0f);
      float rc = rsqrtf(q * (1.0f / 64.0f) - mc * mc + LNEPS);
#pragma unroll
      for (int k = 0; k < 2; ++k) {
        int j  = lane * 2 + k;
        float ln = (cv[k] - mc) * rc * gc_g[j] + gc_b[j];
        float h  = sigmoidf(ov[k]) * tanh_fast(ln);
        u16 hb = to_bf16(h);
        sH[r][j] = hb;                                   // feed next step
        hs[(long)(rowb + r) * LH + t * H_SZ + j] = hb;   // out_seq (bf16)
      }
    }
    __syncthreads();
    ax = axn;
  }
}

// ---------------- output projection: [B,9600] x [9600,150] ----------------
// one wave per 16x16 output tile; K = 9600 -> 300 bf16 WMMAs split over
// 4 independent accumulators to break the WMMA->WMMA RAW chain (4-way ILP)
__global__ __launch_bounds__(32) void k_outgemm(
    const u16* __restrict__ hs, const u16* __restrict__ wout,
    const float* __restrict__ bout, float* __restrict__ out) {
  const int lane  = threadIdx.x & 31;
  const int ntile = blockIdx.x;   // 0..9   (cols, last tile partially masked)
  const int mtile = blockIdx.y;   // 0..127 (batch rows)
  const int nlane = lane & 15;
  const int halfk = lane >> 4;

  int col  = ntile * 16 + nlane;
  int colc = col < NOUT ? col : NOUT - 1;   // clamp; garbage cols never stored
  const u16* ap = hs   + (long)(mtile * 16 + nlane) * LH + halfk * 16;
  const u16* bp = wout + (long)colc * LH + halfk * 16;

  v8f z8 = {0.f, 0.f, 0.f, 0.f, 0.f, 0.f, 0.f, 0.f};
  v8f acc0 = z8, acc1 = z8, acc2 = z8, acc3 = z8;
  for (int k = 0; k < LH / 32; k += 4) {    // 75 iterations x 4 k-steps
    v16u a0 = *(const v16u*)(ap + (k + 0) * 32);
    v16u b0 = *(const v16u*)(bp + (k + 0) * 32);
    v16u a1 = *(const v16u*)(ap + (k + 1) * 32);
    v16u b1 = *(const v16u*)(bp + (k + 1) * 32);
    v16u a2 = *(const v16u*)(ap + (k + 2) * 32);
    v16u b2 = *(const v16u*)(bp + (k + 2) * 32);
    v16u a3 = *(const v16u*)(ap + (k + 3) * 32);
    v16u b3 = *(const v16u*)(bp + (k + 3) * 32);
    acc0 = wmma_bf16(a0, b0, acc0);
    acc1 = wmma_bf16(a1, b1, acc1);
    acc2 = wmma_bf16(a2, b2, acc2);
    acc3 = wmma_bf16(a3, b3, acc3);
  }
  v8f acc = (acc0 + acc1) + (acc2 + acc3);

  if (col < NOUT) {
    float bb = bout[col];
#pragma unroll
    for (int r = 0; r < 8; ++r) {
      int row = mtile * 16 + halfk * 8 + r;
      out[(long)row * NOUT + col] = acc[r] + bb;
    }
  }
}

// ---------------- host-side launcher ----------------
extern "C" void kernel_launch(void* const* d_in, const int* in_sizes, int n_in,
                              void* d_out, int out_size, void* d_ws, size_t ws_size,
                              hipStream_t stream) {
  const int*   idx   = (const int*)  d_in[0];
  const float* eps   = (const float*)d_in[1];
  const float* h0    = (const float*)d_in[2];
  const float* c0    = (const float*)d_in[3];
  const float* mt    = (const float*)d_in[4];
  const float* vt    = (const float*)d_in[5];
  const float* Wih   = (const float*)d_in[6];
  const float* Whh   = (const float*)d_in[7];
  const float* gih_g = (const float*)d_in[8];
  const float* gih_b = (const float*)d_in[9];
  const float* ghh_g = (const float*)d_in[10];
  const float* ghh_b = (const float*)d_in[11];
  const float* gc_g  = (const float*)d_in[12];
  const float* gc_b  = (const float*)d_in[13];
  const float* Wout  = (const float*)d_in[14];
  const float* bout  = (const float*)d_in[15];
  float* out = (float*)d_out;

  char* p = (char*)d_ws;
  auto carve = [&](size_t bytes) -> char* {
    char* r = p;
    p += (bytes + 255) & ~(size_t)255;
    return r;
  };
  u16* rep   = (u16*)carve((size_t)B_SZ * L_SZ * E_SZ * 2);   // ~9.8 MB
  u16* hsbuf = (u16*)carve((size_t)B_SZ * LH * 2);            // ~39.3 MB
  u16* wihb  = (u16*)carve((size_t)G_SZ * E_SZ * 2);
  u16* whhb  = (u16*)carve((size_t)G_SZ * H_SZ * 2);
  u16* woutb = (u16*)carve((size_t)NOUT * LH * 2);            // ~2.9 MB

  k_cvt_bf16<<<(G_SZ * E_SZ + 255) / 256, 256, 0, stream>>>(Wih, wihb, G_SZ * E_SZ);
  k_cvt_bf16<<<(G_SZ * H_SZ + 255) / 256, 256, 0, stream>>>(Whh, whhb, G_SZ * H_SZ);
  k_cvt_bf16<<<(NOUT * LH + 255) / 256, 256, 0, stream>>>(Wout, woutb, NOUT * LH);
  k_reparam<<<(B_SZ * L_SZ + 255) / 256, 256, 0, stream>>>(idx, eps, mt, vt, rep);

  k_lstm<<<B_SZ / 16, 256, 0, stream>>>(rep, wihb, whhb, h0, c0,
                                        gih_g, gih_b, ghh_g, ghh_b, gc_g, gc_b,
                                        hsbuf);

  k_outgemm<<<dim3((NOUT + 15) / 16, B_SZ / 16), 32, 0, stream>>>(hsbuf, woutb, bout, out);
}
